// rnn_46514495816222
// MI455X (gfx1250) — matrix operand order repaired and checked by reading, not compiled, not run on hardware
//
#include <hip/hip_runtime.h>
#include <hip/hip_bf16.h>

// ---------------------------------------------------------------------------
// Types for CDNA5 WMMA (wave32): D(16x16 f32) = A(16x32 bf16) x B(32x16 bf16) + C
// ---------------------------------------------------------------------------
typedef __attribute__((ext_vector_type(16))) __bf16 v16bf;
typedef __attribute__((ext_vector_type(8)))  __bf16 v8bf;
typedef __attribute__((ext_vector_type(8)))  float  v8f;
typedef int v4i __attribute__((vector_size(16)));   // matches builtin's V4i param

#define AS1 __attribute__((address_space(1)))
#define AS3 __attribute__((address_space(3)))

#if defined(__has_builtin)
#if __has_builtin(__builtin_amdgcn_global_load_async_to_lds_b128)
#define HAVE_ASYNC_LDS 1
#else
#define HAVE_ASYNC_LDS 0
#endif
#else
#define HAVE_ASYNC_LDS 0
#endif

#if HAVE_ASYNC_LDS
#if __has_builtin(__builtin_amdgcn_s_wait_asynccnt)
#define WAIT_ASYNC0() __builtin_amdgcn_s_wait_asynccnt(0)
#else
#define WAIT_ASYNC0() asm volatile("s_wait_asynccnt 0" ::: "memory")
#endif
#else
#define WAIT_ASYNC0() ((void)0)
#endif

#define TAU 0.25f
#define OMT 0.75f
#define BATCH 1024
#define TSTEPS 12
#define IN_DIM 1024
#define HID 1024
#define OUT_DIM 1024
#define CLN 512

// Block tile: 128(M) x 128(N), K staged in chunks of 64 (double buffered LDS).
// 8 waves as 4(M) x 2(N); each wave owns a 32x64 strip -> acc[2][4] 16x16 tiles.
#define BM 128
#define BN 128
#define BK 64

// ---------------------------------------------------------------------------
// Stage one 128x64 bf16 tile (rows of A, or N-rows of Bt) into LDS.
// 128 rows x 64 bf16 = 1024 x 16B chunks; 256 threads x 4 chunks each.
// Uses CDNA5 async global->LDS copies (ASYNCcnt) when available.
// ---------------------------------------------------------------------------
__device__ __forceinline__ void copy_tile(const __bf16* __restrict__ g, int ldK,
                                          int row0, int kblk,
                                          __bf16* __restrict__ lds, int tid)
{
#pragma unroll
  for (int j = 0; j < 4; ++j) {
    const int chunk = tid + 256 * j;           // 0..1023, coalesced across threads
    const int row = chunk >> 3;                // 8 chunks per 64-elem row
    const int c8  = (chunk & 7) << 3;
    const __bf16* gp = g + (size_t)(row0 + row) * ldK + kblk + c8;
    __bf16* lp = lds + row * BK + c8;
#if HAVE_ASYNC_LDS
    __builtin_amdgcn_global_load_async_to_lds_b128((AS1 v4i*)gp, (AS3 v4i*)lp, 0, 0);
#else
    *(v8bf*)lp = *(const v8bf*)gp;
#endif
  }
}

// ---------------------------------------------------------------------------
// One K=64 stage of WMMA from LDS.  Fragment layouts per ISA 7.12.2:
//   A 16x32: lanes 0-15 row l: K 0..7 / 16..23; lanes 16-31: K 8..15 / 24..31.
//   B 32x16: lanes 0-15 col l: K 0..7 / 16..23; lanes 16-31: K 8..15 / 24..31.
// ---------------------------------------------------------------------------
__device__ __forceinline__ void compute_stage(const __bf16* __restrict__ lA,
                                              const __bf16* __restrict__ lB,
                                              int waveM, int waveN, int l, int hi,
                                              v8f acc[2][4])
{
#pragma unroll
  for (int ks = 0; ks < 2; ++ks) {
    const int kb = ks * 32;
    v16bf a[2];
#pragma unroll
    for (int mi = 0; mi < 2; ++mi) {
      const __bf16* ar = lA + (waveM * 32 + mi * 16 + l) * BK + kb + hi * 8;
      ((v8bf*)&a[mi])[0] = *(const v8bf*)(ar);
      ((v8bf*)&a[mi])[1] = *(const v8bf*)(ar + 16);
    }
#pragma unroll
    for (int ni = 0; ni < 4; ++ni) {
      const __bf16* br = lB + (waveN * 64 + ni * 16 + l) * BK + kb + hi * 8;
      v16bf b;
      ((v8bf*)&b)[0] = *(const v8bf*)(br);
      ((v8bf*)&b)[1] = *(const v8bf*)(br + 16);
#pragma unroll
      for (int mi = 0; mi < 2; ++mi)
        acc[mi][ni] = __builtin_amdgcn_wmma_f32_16x16x32_bf16(
            false, a[mi], false, b, (short)0, acc[mi][ni], false, false);
    }
  }
}

// ---------------------------------------------------------------------------
// Full staged block GEMM: C[128x128] += A[m0g:+128, :K] * Bt[n0g:+128, :K]^T
// Double-buffered LDS with async copies overlapping WMMA compute.
// ---------------------------------------------------------------------------
__device__ __forceinline__ void block_gemm(
    const __bf16* __restrict__ A, const __bf16* __restrict__ Bt, int K,
    int m0g, int n0g,
    __bf16* lA0, __bf16* lA1, __bf16* lB0, __bf16* lB1,
    int tid, int waveM, int waveN, int l, int hi, v8f acc[2][4])
{
  const int NB = K >> 6;                       // stages of 64
  copy_tile(A,  K, m0g, 0, lA0, tid);
  copy_tile(Bt, K, n0g, 0, lB0, tid);
  WAIT_ASYNC0();
  __syncthreads();
  for (int kb = 0; kb < NB; ++kb) {
    __bf16* curA = (kb & 1) ? lA1 : lA0;
    __bf16* curB = (kb & 1) ? lB1 : lB0;
    if (kb + 1 < NB) {                         // prefetch next stage into other buf
      __bf16* nxtA = (kb & 1) ? lA0 : lA1;
      __bf16* nxtB = (kb & 1) ? lB0 : lB1;
      copy_tile(A,  K, m0g, (kb + 1) * BK, nxtA, tid);
      copy_tile(Bt, K, n0g, (kb + 1) * BK, nxtB, tid);
    }
    compute_stage(curA, curB, waveM, waveN, l, hi, acc);
    WAIT_ASYNC0();                             // my async copies for kb+1 landed
    __syncthreads();                           // all waves done with stage kb
  }
}

// out = TAU*(acc + bias) + (1-TAU)*prev over the wave's 32x64 strip
__device__ __forceinline__ void wave_epilogue(
    const v8f acc[2][4], const float* __restrict__ prev, float* __restrict__ outp,
    const float* __restrict__ bias, int ldN, int m0g, int n0g,
    int waveM, int waveN, int l, int hi)
{
#pragma unroll
  for (int ni = 0; ni < 4; ++ni) {
    const int n  = n0g + waveN * 64 + ni * 16 + l;
    const float bv = bias[n];
#pragma unroll
    for (int mi = 0; mi < 2; ++mi) {
#pragma unroll
      for (int r = 0; r < 8; ++r) {
        const size_t idx =
            (size_t)(m0g + waveM * 32 + mi * 16 + r + 8 * hi) * ldN + n;
        outp[idx] = TAU * acc[mi][ni][r] + TAU * bv + OMT * prev[idx];
      }
    }
  }
}

// ---------------------------------------------------------------------------
// Fused per-timestep GEMM kernel.  128x128 block tiles:
//   blocks [0,64):    h = TAU*(x @ Woh + bh)  + OMT*h_prev          (8x8)
//   blocks [64,128):  p = TAU*(ah@Whp + ap@Wpp + ac@Wcp + bp) + OMT*p_prev
//   blocks [128,160): c = TAU*(ap@Wpc + bc)   + OMT*c_prev          (8x4)
// ---------------------------------------------------------------------------
__global__ void __launch_bounds__(256)
rnn_step_gemm(const __bf16* __restrict__ xbf,
              const __bf16* __restrict__ a_h,
              const __bf16* __restrict__ a_p,
              const __bf16* __restrict__ a_c,
              const __bf16* __restrict__ wt_oh,   // [1024][1024]  (N,K)
              const __bf16* __restrict__ wt_hp,   // [1024][1024]
              const __bf16* __restrict__ wt_pp,   // [1024][1024]
              const __bf16* __restrict__ wt_cp,   // [1024][512]
              const __bf16* __restrict__ wt_pc,   // [512][1024]
              const float* __restrict__ h_prev, float* __restrict__ h_new,
              const float* __restrict__ p_prev, float* __restrict__ p_new,
              const float* __restrict__ c_prev, float* __restrict__ c_new,
              const float* __restrict__ bias_h,
              const float* __restrict__ bias_p,
              const float* __restrict__ bias_c)
{
  __shared__ __bf16 lA0[BM * BK], lA1[BM * BK];
  __shared__ __bf16 lB0[BN * BK], lB1[BN * BK];

  const int tid   = threadIdx.x;
  const int wid   = tid >> 5;
  const int lane  = tid & 31;
  const int l     = lane & 15;
  const int hi    = lane >> 4;
  const int waveM = wid >> 1;          // 0..3
  const int waveN = wid & 1;           // 0..1

  v8f acc[2][4];
#pragma unroll
  for (int mi = 0; mi < 2; ++mi)
#pragma unroll
    for (int ni = 0; ni < 4; ++ni) acc[mi][ni] = (v8f)(0.0f);

  const int bi = blockIdx.x;
  if (bi < 64) {                       // ---- h path: x @ Woh
    const int m0g = (bi >> 3) * BM;
    const int n0g = (bi & 7) * BN;
    block_gemm(xbf, wt_oh, 1024, m0g, n0g, lA0, lA1, lB0, lB1,
               tid, waveM, waveN, l, hi, acc);
    wave_epilogue(acc, h_prev, h_new, bias_h, 1024, m0g, n0g, waveM, waveN, l, hi);
  } else if (bi < 128) {               // ---- p path: ah@Whp + ap@Wpp + ac@Wcp
    const int i   = bi - 64;
    const int m0g = (i >> 3) * BM;
    const int n0g = (i & 7) * BN;
    block_gemm(a_h, wt_hp, 1024, m0g, n0g, lA0, lA1, lB0, lB1,
               tid, waveM, waveN, l, hi, acc);
    block_gemm(a_p, wt_pp, 1024, m0g, n0g, lA0, lA1, lB0, lB1,
               tid, waveM, waveN, l, hi, acc);
    block_gemm(a_c, wt_cp,  512, m0g, n0g, lA0, lA1, lB0, lB1,
               tid, waveM, waveN, l, hi, acc);
    wave_epilogue(acc, p_prev, p_new, bias_p, 1024, m0g, n0g, waveM, waveN, l, hi);
  } else {                             // ---- c path: ap @ Wpc (N=512)
    const int i   = bi - 128;
    const int m0g = (i >> 2) * BM;
    const int n0g = (i & 3) * BN;
    block_gemm(a_p, wt_pc, 1024, m0g, n0g, lA0, lA1, lB0, lB1,
               tid, waveM, waveN, l, hi, acc);
    wave_epilogue(acc, c_prev, c_new, bias_c, 512, m0g, n0g, waveM, waveN, l, hi);
  }
}

// ---------------------------------------------------------------------------
// Activation + next-input conversion. Runs after each step's GEMM.
// ---------------------------------------------------------------------------
__device__ __forceinline__ float sigf(float x) {
  return 1.0f / (1.0f + __expf(-x));
}

__global__ void __launch_bounds__(256)
rnn_step_act(const float* __restrict__ h, const float* __restrict__ p,
             const float* __restrict__ c,
             __bf16* __restrict__ a_h, __bf16* __restrict__ a_p,
             __bf16* __restrict__ a_c,
             float* __restrict__ out_p,              // nullptr unless tick emitted
             const float* __restrict__ inputs,       // [B][T][I]
             __bf16* __restrict__ xbf_next, int t_next)
{
  const int i = blockIdx.x * blockDim.x + threadIdx.x;
  if (i < BATCH * HID) {
    const float sh = sigf(h[i]);
    a_h[i] = (__bf16)sh;
    const float sp = sigf(p[i]);
    a_p[i] = (__bf16)sp;
    if (out_p) out_p[i] = sp;
    if (t_next < TSTEPS) {
      const int b = i >> 10, col = i & 1023;
      xbf_next[i] = (__bf16)inputs[(size_t)b * TSTEPS * IN_DIM +
                                   (size_t)t_next * IN_DIM + col];
    }
  }
  if (i < BATCH * CLN) {
    a_c[i] = (__bf16)sigf(c[i]);
  }
}

// ---------------------------------------------------------------------------
// Prologue kernels
// ---------------------------------------------------------------------------
__global__ void __launch_bounds__(256)
cvt_transpose(const float* __restrict__ W, __bf16* __restrict__ Wt, int K, int N)
{
  const int i = blockIdx.x * blockDim.x + threadIdx.x;
  if (i >= K * N) return;
  const int k = i / N, n = i % N;
  Wt[(size_t)n * K + k] = (__bf16)W[i];
}

__global__ void __launch_bounds__(256)
rnn_init(float* __restrict__ h0, float* __restrict__ p0, float* __restrict__ c0,
         __bf16* __restrict__ a_h, __bf16* __restrict__ a_p,
         __bf16* __restrict__ a_c,
         const float* __restrict__ inputs, __bf16* __restrict__ xbf0)
{
  const int i = blockIdx.x * blockDim.x + threadIdx.x;
  if (i < BATCH * HID) {
    h0[i] = 0.0f; p0[i] = 0.0f;
    a_h[i] = (__bf16)0.5f; a_p[i] = (__bf16)0.5f;
    const int b = i >> 10, col = i & 1023;
    xbf0[i] = (__bf16)inputs[(size_t)b * TSTEPS * IN_DIM + col];   // t = 0
  }
  if (i < BATCH * CLN) {
    c0[i] = 0.0f;
    a_c[i] = (__bf16)0.5f;
  }
}

// ---------------------------------------------------------------------------
// Host-side launch
// ---------------------------------------------------------------------------
extern "C" void kernel_launch(void* const* d_in, const int* in_sizes, int n_in,
                              void* d_out, int out_size, void* d_ws, size_t ws_size,
                              hipStream_t stream)
{
  (void)in_sizes; (void)n_in; (void)out_size; (void)ws_size;
  const float* inputs = (const float*)d_in[0];   // [B][T][I]
  const float* w_oh   = (const float*)d_in[1];   // [I][H]
  const float* w_hp   = (const float*)d_in[2];   // [H][P]
  const float* w_pp   = (const float*)d_in[3];   // [P][P]
  const float* w_pc   = (const float*)d_in[4];   // [P][C]
  const float* w_cp   = (const float*)d_in[5];   // [C][P]
  const float* bias_h = (const float*)d_in[6];
  const float* bias_p = (const float*)d_in[7];
  const float* bias_c = (const float*)d_in[8];
  float* out = (float*)d_out;                    // [4][B][P]

  // ---- workspace carve-up (all offsets 256B aligned)
  char* base = (char*)d_ws;
  size_t off = 0;
  auto carve = [&](size_t bytes) -> char* {
    char* p = base + off;
    off = (off + bytes + 255) & ~(size_t)255;
    return p;
  };
  __bf16* wt_oh = (__bf16*)carve((size_t)IN_DIM * HID * 2);      // [H][I]
  __bf16* wt_hp = (__bf16*)carve((size_t)HID * OUT_DIM * 2);     // [P][H]
  __bf16* wt_pp = (__bf16*)carve((size_t)OUT_DIM * OUT_DIM * 2); // [P][P]
  __bf16* wt_cp = (__bf16*)carve((size_t)CLN * OUT_DIM * 2);     // [P][C]
  __bf16* wt_pc = (__bf16*)carve((size_t)OUT_DIM * CLN * 2);     // [C][P]
  float* hbuf[2] = { (float*)carve((size_t)BATCH * HID * 4),
                     (float*)carve((size_t)BATCH * HID * 4) };
  float* pbuf[2] = { (float*)carve((size_t)BATCH * OUT_DIM * 4),
                     (float*)carve((size_t)BATCH * OUT_DIM * 4) };
  float* cbuf[2] = { (float*)carve((size_t)BATCH * CLN * 4),
                     (float*)carve((size_t)BATCH * CLN * 4) };
  __bf16* a_h = (__bf16*)carve((size_t)BATCH * HID * 2);
  __bf16* a_p = (__bf16*)carve((size_t)BATCH * OUT_DIM * 2);
  __bf16* a_c = (__bf16*)carve((size_t)BATCH * CLN * 2);
  __bf16* xbf = (__bf16*)carve((size_t)BATCH * IN_DIM * 2);

  // ---- prologue: bf16-transpose weights, init state, convert x[:,0,:]
  {
    const int thr = 256;
    cvt_transpose<<<(IN_DIM * HID + thr - 1) / thr, thr, 0, stream>>>(w_oh, wt_oh, IN_DIM, HID);
    cvt_transpose<<<(HID * OUT_DIM + thr - 1) / thr, thr, 0, stream>>>(w_hp, wt_hp, HID, OUT_DIM);
    cvt_transpose<<<(OUT_DIM * OUT_DIM + thr - 1) / thr, thr, 0, stream>>>(w_pp, wt_pp, OUT_DIM, OUT_DIM);
    cvt_transpose<<<(CLN * OUT_DIM + thr - 1) / thr, thr, 0, stream>>>(w_cp, wt_cp, CLN, OUT_DIM);
    cvt_transpose<<<(OUT_DIM * CLN + thr - 1) / thr, thr, 0, stream>>>(w_pc, wt_pc, OUT_DIM, CLN);
    rnn_init<<<(BATCH * HID + thr - 1) / thr, thr, 0, stream>>>(
        hbuf[0], pbuf[0], cbuf[0], a_h, a_p, a_c, inputs, xbf);
  }

  // ---- 12 sequential steps: fused WMMA GEMM + activation
  // 64 (h) + 64 (p) + 32 (c) = 160 block tiles of 128x128
  const int gemm_blocks = 160;
  const int act_blocks  = (BATCH * HID + 255) / 256;
  for (int t = 0; t < TSTEPS; ++t) {
    const int r = t & 1;       // read buffers
    const int w = r ^ 1;       // write buffers
    rnn_step_gemm<<<gemm_blocks, 256, 0, stream>>>(
        xbf, a_h, a_p, a_c,
        wt_oh, wt_hp, wt_pp, wt_cp, wt_pc,
        hbuf[r], hbuf[w], pbuf[r], pbuf[w], cbuf[r], cbuf[w],
        bias_h, bias_p, bias_c);
    float* out_p = (t >= TSTEPS - 4)
                   ? out + (size_t)(t - (TSTEPS - 4)) * BATCH * OUT_DIM
                   : (float*)nullptr;
    rnn_step_act<<<act_blocks, 256, 0, stream>>>(
        hbuf[w], pbuf[w], cbuf[w], a_h, a_p, a_c,
        out_p, inputs, xbf, t + 1);
  }
}
